// VarianceAdaptor_68925635166240
// MI455X (gfx1250) — matrix operand order repaired and checked by reading, not compiled, not run on hardware
//
#include <hip/hip_runtime.h>
#include <hip/hip_bf16.h>

typedef __bf16 v16bf __attribute__((ext_vector_type(16)));
typedef float  v8f   __attribute__((ext_vector_type(8)));

// ---------- bf16 helpers (round-to-nearest) ----------
__device__ inline unsigned short f2bf(float f) {
    unsigned u = __float_as_uint(f);
    unsigned r = u + 0x7FFFu + ((u >> 16) & 1u);
    return (unsigned short)(r >> 16);
}
__device__ inline float bf2f(unsigned short h) {
    return __uint_as_float(((unsigned)h) << 16);
}
__device__ inline float ldf(const float* p, size_t i) { return p[i]; }
__device__ inline float ldf(const unsigned short* p, size_t i) { return bf2f(p[i]); }

// ============================================================================
// Fused conv1d(k=3,pad=1) as WMMA-bf16 GEMM + bias + LayerNorm + ReLU epilogue.
// in : [B, T, 256] (float32 or bf16) ; w : [256, 256, 3] f32
// out: [B, T, 256] bf16
// grid = (T/64, B), block = 256 threads (8 wave32).
// Block tile: 64 time rows x 256 features.
//   Wave wv owns N columns {32wv..32wv+31}; 4 M sub-tiles of 16 rows each.
//   Accumulators: acc[mt][0/1] = v8f, C layout row m = 16*mt + r + 8*half,
//   col f = 32*wv + l15 (+16 for the second N tile).
// LayerNorm stats computed from registers (shfl within 16-lane half +
// small LDS partials), output written bf16 straight from registers.
// ============================================================================
template <typename InT>
__global__ __launch_bounds__(256)
void conv_ln_relu_kernel(const InT* __restrict__ in,
                         const float* __restrict__ w,
                         const float* __restrict__ bias,
                         const float* __restrict__ gamma,
                         const float* __restrict__ beta,
                         unsigned short* __restrict__ out,
                         int T)
{
    constexpr int D = 256;
    const int b    = blockIdx.y;
    const int t0   = blockIdx.x * 64;
    const int tid  = threadIdx.x;
    const int wv   = tid >> 5;      // wave 0..7
    const int lane = tid & 31;
    const int half = lane >> 4;     // 0/1
    const int l15  = lane & 15;

    __shared__ unsigned short lA[64][32];      // t-rows x K-chunk (bf16)  4 KB
    __shared__ unsigned short lB[256][32];     // f x K-chunk      (bf16) 16 KB
    __shared__ float pSum[64][8];              // per-row per-wave partials 2 KB
    __shared__ float pSq[64][8];               //                           2 KB
    __shared__ float sMean[64];
    __shared__ float sInv[64];

    v8f acc[4][2];
    #pragma unroll
    for (int mt = 0; mt < 4; ++mt) { acc[mt][0] = (v8f){}; acc[mt][1] = (v8f){}; }

    for (int kc = 0; kc < 3; ++kc) {
        for (int d0 = 0; d0 < D; d0 += 32) {
            // ---- stage A tile: x[b, t0+tt+kc-1, d0+kk] -> bf16 (8/thread) ----
            #pragma unroll
            for (int i = tid; i < 64 * 32; i += 256) {
                int tt = i >> 5, kk = i & 31;
                int tr = t0 + tt + kc - 1;
                float v = 0.f;
                if (tr >= 0 && tr < T)
                    v = ldf(in, ((size_t)b * T + tr) * D + d0 + kk);
                lA[tt][kk] = f2bf(v);
            }
            // ---- stage B tile: w[f=tid, d0+kk, kc] -> bf16 (32/thread) ----
            {
                const float* wp = w + (size_t)tid * (D * 3) + (size_t)d0 * 3 + kc;
                __builtin_prefetch((const void*)(wp + 96), 0, 0); // next d-chunk
                #pragma unroll 8
                for (int kk = 0; kk < 32; ++kk)
                    lB[tid][kk] = f2bf(wp[kk * 3]);
            }
            __syncthreads();

            // ---- fragments: contiguous k = 16*half + e, same order for A and B ----
            union U16 { unsigned short u[16]; v16bf v; };
            U16 bf0, bf1;
            #pragma unroll
            for (int e = 0; e < 16; ++e) {
                int kb = 16 * half + e;                    // B 32x16: col=l15
                bf0.u[e] = lB[32 * wv + l15][kb];
                bf1.u[e] = lB[32 * wv + 16 + l15][kb];
            }
            #pragma unroll
            for (int mt = 0; mt < 4; ++mt) {
                U16 af;
                #pragma unroll
                for (int e = 0; e < 16; ++e) {
                    // A 16x32: row = l15; K = 16*half+e
                    int ka = 16 * half + e;
                    af.u[e] = lA[mt * 16 + l15][ka];
                }
                acc[mt][0] = __builtin_amdgcn_wmma_f32_16x16x32_bf16(
                                 false, af.v, false, bf0.v, (short)0, acc[mt][0], false, false);
                acc[mt][1] = __builtin_amdgcn_wmma_f32_16x16x32_bf16(
                                 false, af.v, false, bf1.v, (short)0, acc[mt][1], false, false);
            }
            __syncthreads();
        }
    }

    // ---- bias ----
    const int f0 = 32 * wv + l15;
    const int f1 = f0 + 16;
    const float bias0 = bias[f0], bias1 = bias[f1];
    #pragma unroll
    for (int mt = 0; mt < 4; ++mt)
        #pragma unroll
        for (int r = 0; r < 8; ++r) {
            acc[mt][0][r] += bias0;
            acc[mt][1][r] += bias1;
        }

    // ---- per-row partial sums (this wave's 32 columns) ----
    #pragma unroll
    for (int mt = 0; mt < 4; ++mt)
        #pragma unroll
        for (int r = 0; r < 8; ++r) {
            float v0 = acc[mt][0][r], v1 = acc[mt][1][r];
            float s = v0 + v1;
            float q = v0 * v0 + v1 * v1;
            #pragma unroll
            for (int off = 1; off < 16; off <<= 1) {   // reduce within 16-lane half
                s += __shfl_xor(s, off, 32);
                q += __shfl_xor(q, off, 32);
            }
            int m = mt * 16 + r + 8 * half;
            if (l15 == 0) { pSum[m][wv] = s; pSq[m][wv] = q; }
        }
    __syncthreads();

    // ---- finalize stats: one thread per row ----
    if (tid < 64) {
        float s = 0.f, q = 0.f;
        #pragma unroll
        for (int j = 0; j < 8; ++j) { s += pSum[tid][j]; q += pSq[tid][j]; }
        float mean = s * (1.f / 256.f);
        float var  = q * (1.f / 256.f) - mean * mean;
        sMean[tid] = mean;
        sInv[tid]  = rsqrtf(var + 1e-5f);
    }
    __syncthreads();

    // ---- normalize + ReLU + store bf16 straight from registers ----
    const float g0 = gamma[f0], g1 = gamma[f1];
    const float be0 = beta[f0], be1 = beta[f1];
    #pragma unroll
    for (int mt = 0; mt < 4; ++mt)
        #pragma unroll
        for (int r = 0; r < 8; ++r) {
            int m = mt * 16 + r + 8 * half;
            float mean = sMean[m], inv = sInv[m];
            float v0 = (acc[mt][0][r] - mean) * inv * g0 + be0;
            float v1 = (acc[mt][1][r] - mean) * inv * g1 + be1;
            v0 = v0 > 0.f ? v0 : 0.f;
            v1 = v1 > 0.f ? v1 : 0.f;
            size_t base = ((size_t)b * T + t0 + m) * 256;
            out[base + f0] = f2bf(v0);
            out[base + f1] = f2bf(v1);
        }
}

// ---------------- final linear (256 -> 1) + ReLU -----------------------------
__global__ __launch_bounds__(256)
void final_linear_relu(const unsigned short* __restrict__ h,
                       const float* __restrict__ wl,
                       const float* __restrict__ bl,
                       float* __restrict__ out)
{
    int row  = blockIdx.x * 8 + (threadIdx.x >> 5);   // b*T + t
    int lane = threadIdx.x & 31;
    const unsigned short* p = h + (size_t)row * 256;
    float s = 0.f;
    #pragma unroll
    for (int j = lane; j < 256; j += 32) s += bf2f(p[j]) * wl[j];
    #pragma unroll
    for (int off = 16; off; off >>= 1) s += __shfl_xor(s, off, 32);
    if (lane == 0) {
        float v = s + bl[0];
        out[row] = v > 0.f ? v : 0.f;
    }
}

// ---------------- inclusive cumsum of durations (Tin=256) --------------------
__global__ __launch_bounds__(256)
void cumsum_kernel(const int* __restrict__ dur, int* __restrict__ cum)
{
    __shared__ int s[256];
    int b = blockIdx.x, t = threadIdx.x;
    s[t] = dur[b * 256 + t];
    __syncthreads();
    for (int off = 1; off < 256; off <<= 1) {
        int v = (t >= off) ? s[t - off] : 0;
        __syncthreads();
        s[t] += v;
        __syncthreads();
    }
    cum[b * 256 + t] = s[t];
}

// ---------------- length regulate: gather rows via searchsorted(right) -------
__global__ __launch_bounds__(256)
void length_regulate_kernel(const float* __restrict__ x,
                            const int* __restrict__ cum,
                            float* __restrict__ xe,
                            int Tin, int mel)
{
    int row  = blockIdx.x * 8 + (threadIdx.x >> 5);   // b*mel + t
    int lane = threadIdx.x & 31;
    int b = row / mel, t = row % mel;
    const int* c  = cum + b * Tin;
    int total = c[Tin - 1];
    int idx = 0;
    if (lane == 0) {
        int lo = 0, hi = Tin;                 // first i with c[i] > t
        while (lo < hi) {
            int mid = (lo + hi) >> 1;
            if (c[mid] <= t) lo = mid + 1; else hi = mid;
        }
        idx = lo < Tin ? lo : Tin - 1;        // clip
    }
    idx = __shfl(idx, 0, 32);
    bool valid = t < total;
    const float4* src = (const float4*)(x + ((size_t)b * Tin + idx) * 256);
    float4* dst = (float4*)(xe + (size_t)row * 256);
    #pragma unroll
    for (int j = lane; j < 64; j += 32)
        dst[j] = valid ? src[j] : make_float4(0.f, 0.f, 0.f, 0.f);
}

// ---------------- xe += emb[searchsorted(bins, target, left)] ----------------
__global__ __launch_bounds__(256)
void add_emb_kernel(float* __restrict__ xe,
                    const float* __restrict__ bins,
                    const float* __restrict__ target,
                    const float* __restrict__ emb)
{
    int row  = blockIdx.x * 8 + (threadIdx.x >> 5);   // b*mel + t
    int lane = threadIdx.x & 31;
    float tv = target[row];
    int idx = 0;
    if (lane == 0) {
        int lo = 0, hi = 255;                 // first i with bins[i] >= tv
        while (lo < hi) {
            int mid = (lo + hi) >> 1;
            if (bins[mid] < tv) lo = mid + 1; else hi = mid;
        }
        idx = lo;
    }
    idx = __shfl(idx, 0, 32);
    const float4* e = (const float4*)(emb + (size_t)idx * 256);
    float4* dst = (float4*)(xe + (size_t)row * 256);
    #pragma unroll
    for (int j = lane; j < 64; j += 32) {
        float4 v = dst[j], ev = e[j];
        v.x += ev.x; v.y += ev.y; v.z += ev.z; v.w += ev.w;
        dst[j] = v;
    }
}

// ============================================================================
extern "C" void kernel_launch(void* const* d_in, const int* in_sizes, int n_in,
                              void* d_out, int out_size, void* d_ws, size_t ws_size,
                              hipStream_t stream)
{
    (void)in_sizes; (void)n_in; (void)out_size; (void)ws_size;
    constexpr int B = 32, TIN = 256, MEL = 2048;

    auto F32 = [&](int i) { return (const float*)d_in[i]; };

    const float* x        = F32(0);
    const int*   dur      = (const int*)d_in[1];
    const float* pitch_t  = F32(2);
    const float* energy_t = F32(3);
    // d_in[4] = mel_max_length scalar (hardcoded 2048)
    // param packs: {w1,b1,g1,be1,w2,b2,g2,be2,wl,bl}
    const int DP = 5, PP = 15, EP = 25;
    const float* pitch_bins  = F32(35);
    const float* energy_bins = F32(36);
    const float* pitch_emb   = F32(37);
    const float* energy_emb  = F32(38);

    // ---- workspace layout ----
    char* ws = (char*)d_ws;
    int* cum = (int*)ws;                                           // 32 KB
    unsigned short* h1 = (unsigned short*)(ws + (1u << 20));       // 32 MB
    unsigned short* h2 = h1 + (size_t)B * MEL * 256;               // 32 MB

    // ---- output layout ----
    float* xe     = (float*)d_out;
    float* logdur = xe + (size_t)B * MEL * 256;
    float* ppred  = logdur + (size_t)B * TIN;
    float* epred  = ppred + (size_t)B * MEL;

    dim3 blk(256);

    // ---- 1. duration predictor on x (T = 256) ----
    conv_ln_relu_kernel<float><<<dim3(TIN / 64, B), blk, 0, stream>>>(
        x, F32(DP + 0), F32(DP + 1), F32(DP + 2), F32(DP + 3), h1, TIN);
    conv_ln_relu_kernel<unsigned short><<<dim3(TIN / 64, B), blk, 0, stream>>>(
        h1, F32(DP + 4), F32(DP + 5), F32(DP + 6), F32(DP + 7), h2, TIN);
    final_linear_relu<<<(B * TIN) / 8, blk, 0, stream>>>(h2, F32(DP + 8), F32(DP + 9), logdur);

    // ---- 2. length regulate ----
    cumsum_kernel<<<B, blk, 0, stream>>>(dur, cum);
    length_regulate_kernel<<<(B * MEL) / 8, blk, 0, stream>>>(x, cum, xe, TIN, MEL);

    // ---- 3. pitch predictor on xe (T = 2048) ----
    conv_ln_relu_kernel<float><<<dim3(MEL / 64, B), blk, 0, stream>>>(
        xe, F32(PP + 0), F32(PP + 1), F32(PP + 2), F32(PP + 3), h1, MEL);
    conv_ln_relu_kernel<unsigned short><<<dim3(MEL / 64, B), blk, 0, stream>>>(
        h1, F32(PP + 4), F32(PP + 5), F32(PP + 6), F32(PP + 7), h2, MEL);
    final_linear_relu<<<(B * MEL) / 8, blk, 0, stream>>>(h2, F32(PP + 8), F32(PP + 9), ppred);

    // ---- 4. xe += pitch_emb[bucketize(pitch_target)] ----
    add_emb_kernel<<<(B * MEL) / 8, blk, 0, stream>>>(xe, pitch_bins, pitch_t, pitch_emb);

    // ---- 5. energy predictor on updated xe ----
    conv_ln_relu_kernel<float><<<dim3(MEL / 64, B), blk, 0, stream>>>(
        xe, F32(EP + 0), F32(EP + 1), F32(EP + 2), F32(EP + 3), h1, MEL);
    conv_ln_relu_kernel<unsigned short><<<dim3(MEL / 64, B), blk, 0, stream>>>(
        h1, F32(EP + 4), F32(EP + 5), F32(EP + 6), F32(EP + 7), h2, MEL);
    final_linear_relu<<<(B * MEL) / 8, blk, 0, stream>>>(h2, F32(EP + 8), F32(EP + 9), epred);

    // ---- 6. xe += energy_emb[bucketize(energy_target)] ----
    add_emb_kernel<<<(B * MEL) / 8, blk, 0, stream>>>(xe, energy_bins, energy_t, energy_emb);
}
